// SimpleSelfAttention_48077863911671
// MI455X (gfx1250) — matrix operand order repaired and checked by reading, not compiled, not run on hardware
//
#include <hip/hip_runtime.h>

// SimpleSelfAttention for MI455X (gfx1250): bf16 WMMA flash attention,
// transposed formulation (S^T = K Q^T, O^T = V^T P^T) with double-buffered
// GLOBAL_LOAD_ASYNC_TO_LDS_B128 staging (ASYNCcnt) overlapping the WMMA loop.

#define IN_CH  256
#define OUT_CH 256
#define DKQ    32
#define HN     4096
#define NB     4
#define LROW   40   // padded LDS row stride in halves (80B: 16B-aligned, bank-spread)
#define NCH    (HN / 32)

typedef __attribute__((ext_vector_type(16))) __bf16 v16bf;
typedef __attribute__((ext_vector_type(8)))  float  v8f;

union BF16x16 { v16bf v; uint4 u[2]; };

__device__ __forceinline__ unsigned short f32_to_bf16(float f) {
    unsigned int u = __float_as_uint(f);
    u += 0x7FFFu + ((u >> 16) & 1u);           // round to nearest even
    return (unsigned short)(u >> 16);
}

// ---------------------------------------------------------------- kernel 1
// Q/K projection: per-pixel dot products; weight loads are wave-uniform
// (scalar path). Stores bf16, Q pre-scaled by 1/sqrt(DK).
__global__ __launch_bounds__(256) void proj_qk_kernel(
    const float* __restrict__ x,
    const float* __restrict__ wq, const float* __restrict__ bq,
    const float* __restrict__ wk, const float* __restrict__ bk,
    unsigned short* __restrict__ Qg, unsigned short* __restrict__ Kg)
{
    const int pix = blockIdx.x * 256 + threadIdx.x;   // 0 .. NB*HN-1
    const int b = pix >> 12;
    const int n = pix & (HN - 1);

    float qa[DKQ], ka[DKQ];
#pragma unroll
    for (int d = 0; d < DKQ; ++d) { qa[d] = 0.f; ka[d] = 0.f; }

    const float* xp = x + ((size_t)b * IN_CH) * HN + n;
    for (int c = 0; c < IN_CH; ++c) {
        float xv = xp[(size_t)c * HN];
#pragma unroll
        for (int d = 0; d < DKQ; ++d) {
            qa[d] = fmaf(xv, wq[d * IN_CH + c], qa[d]);
            ka[d] = fmaf(xv, wk[d * IN_CH + c], ka[d]);
        }
    }
    const float scale = 0.17677669529663687f;  // 1/sqrt(32)
    alignas(16) unsigned short qh[DKQ];
    alignas(16) unsigned short kh[DKQ];
#pragma unroll
    for (int d = 0; d < DKQ; ++d) {
        qh[d] = f32_to_bf16((qa[d] + bq[d]) * scale);
        kh[d] = f32_to_bf16(ka[d] + bk[d]);
    }
    uint4* qdst = (uint4*)(Qg + (size_t)pix * DKQ);
    uint4* kdst = (uint4*)(Kg + (size_t)pix * DKQ);
#pragma unroll
    for (int i = 0; i < 4; ++i) {
        qdst[i] = ((const uint4*)qh)[i];
        kdst[i] = ((const uint4*)kh)[i];
    }
}

// ---------------------------------------------------------------- kernel 2
// V projection -> Vt[b][o][n] (o-major so attention stages [o][m] tiles with
// row-contiguous copies). One wave = one 32-channel group (uniform weights).
__global__ __launch_bounds__(256) void proj_v_kernel(
    const float* __restrict__ x,
    const float* __restrict__ wv, const float* __restrict__ bv,
    unsigned short* __restrict__ Vt)
{
    __shared__ float xs[IN_CH * 32];            // [c][n] 32 KB
    const int b  = blockIdx.x >> 7;             // 128 blocks per batch
    const int n0 = (blockIdx.x & 127) * 32;

    for (int idx = threadIdx.x; idx < IN_CH * 32; idx += 256) {
        int c = idx >> 5, nl = idx & 31;
        xs[idx] = x[((size_t)b * IN_CH + c) * HN + n0 + nl];
    }
    __syncthreads();

    const int og = threadIdx.x >> 5;            // wave id = channel group
    const int nl = threadIdx.x & 31;
    const int n  = n0 + nl;
    float acc[32];
#pragma unroll
    for (int o = 0; o < 32; ++o) acc[o] = 0.f;
    const float* wrow = wv + (size_t)(og * 32) * IN_CH;
    for (int c = 0; c < IN_CH; ++c) {
        float xv = xs[c * 32 + nl];
#pragma unroll
        for (int o = 0; o < 32; ++o)
            acc[o] = fmaf(xv, wrow[o * IN_CH + c], acc[o]);
    }
#pragma unroll
    for (int o = 0; o < 32; ++o) {
        int oc = og * 32 + o;
        Vt[((size_t)b * OUT_CH + oc) * HN + n] = f32_to_bf16(acc[o] + bv[oc]);
    }
}

// ---------------------------------------------------------------- kernel 3
// Transposed flash attention. Block = 128 threads (4 waves), 64 query rows
// per block (query row i == lane&15 column of every tile). K/V chunks are
// staged with double-buffered global_load_async_to_lds_b128 (9 wave-level
// async copies per chunk) overlapped with the previous chunk's 18 WMMAs.
__global__ __launch_bounds__(128) void flash_attn_kernel(
    const unsigned short* __restrict__ Qg, const unsigned short* __restrict__ Kg,
    const unsigned short* __restrict__ Vt, const float* __restrict__ x,
    const float* __restrict__ gamma, float* __restrict__ out)
{
    alignas(16) __shared__ unsigned short ldsK[2][32 * LROW];      // [j][d], padded
    alignas(16) __shared__ unsigned short ldsV[2][OUT_CH * LROW];  // [o][m], padded
    alignas(16) __shared__ unsigned short ldsP[4][16 * LROW];      // per-wave P^T [i][m]

    const int b    = blockIdx.x >> 6;
    const int tile = blockIdx.x & 63;
    const int wave = threadIdx.x >> 5;
    const int lane = threadIdx.x & 31;
    const int h    = lane >> 4;                     // wave half
    const int ln   = lane & 15;
    const int iw   = tile * 64 + wave * 16;         // first query row of this wave
    const int tid  = threadIdx.x;

    const unsigned short* Kbase = Kg + (size_t)(b * HN) * DKQ;
    const unsigned short* Vbase = Vt + (size_t)(b * OUT_CH) * HN;

    // per-thread async staging coordinates (block covers 32x32 K + 256x32 V)
    const int kj = tid >> 2, kq = tid & 3;          // K: row, 16B-quad

    // Issue one chunk's staging: 1 K b128 + 8 V b128 wave-level async copies.
    auto stage_async = [&](int j0, int buf) {
        unsigned kl = (unsigned)(uintptr_t)(&ldsK[buf][kj * LROW]) + (unsigned)(kq * 16);
        unsigned kvo = (unsigned)(((j0 + kj) * DKQ + kq * 8) * 2);
        asm volatile("global_load_async_to_lds_b128 %0, %1, %2"
                     :: "v"(kl), "v"(kvo), "s"(Kbase) : "memory");
#pragma unroll
        for (int k = 0; k < 8; ++k) {
            int idx2 = tid + k * 128;
            int o = idx2 >> 2, q = idx2 & 3;
            unsigned vl = (unsigned)(uintptr_t)(&ldsV[buf][o * LROW]) + (unsigned)(q * 16);
            unsigned vvo = (unsigned)((o * HN + j0 + q * 8) * 2);
            asm volatile("global_load_async_to_lds_b128 %0, %1, %2"
                         :: "v"(vl), "v"(vvo), "s"(Vbase) : "memory");
        }
    };

    // Q^T B-operand: lane = column i (query row iw+ln), d = 8h+s / 16+8h+(s-8)
    BF16x16 qb;
    {
        const unsigned short* qrow = Qg + ((size_t)(b * HN) + iw + ln) * DKQ;
        qb.u[0] = *(const uint4*)(qrow + h * 8);
        qb.u[1] = *(const uint4*)(qrow + 16 + h * 8);
    }

    v8f zacc = {};
    v8f Ot[16];                                     // O^T: tile t rows o=t*16+r+8h, col i=ln
#pragma unroll
    for (int t = 0; t < 16; ++t) Ot[t] = zacc;
    float mrun = -3.0e38f, lrun = 0.f;              // per query row (lane-local)

    unsigned short* Pw = ldsP[wave];

    stage_async(0, 0);                              // prologue: chunk 0 in flight

    for (int c = 0; c < NCH; ++c) {
        const int buf = c & 1;
        const unsigned short* bK = ldsK[buf];
        const unsigned short* bV = ldsV[buf];

        asm volatile("s_wait_asynccnt 0x0" ::: "memory");  // my chunk-c copies done
        __syncthreads();                                   // everyone's copies done
        if (c + 1 < NCH) stage_async((c + 1) * 32, buf ^ 1);

        // --- S^T = K_tile @ Q^T (two 16-row key tiles, operands preloaded) ---
        BF16x16 ka0, ka1;
        {
            const unsigned short* kr0 = bK + ln * LROW;
            const unsigned short* kr1 = bK + (16 + ln) * LROW;
            ka0.u[0] = *(const uint4*)(kr0 + h * 8);
            ka0.u[1] = *(const uint4*)(kr0 + 16 + h * 8);
            ka1.u[0] = *(const uint4*)(kr1 + h * 8);
            ka1.u[1] = *(const uint4*)(kr1 + 16 + h * 8);
        }
        v8f S[2];
        S[0] = __builtin_amdgcn_wmma_f32_16x16x32_bf16(
            false, ka0.v, false, qb.v, (short)0, zacc, false, false);
        S[1] = __builtin_amdgcn_wmma_f32_16x16x32_bf16(
            false, ka1.v, false, qb.v, (short)0, zacc, false, false);

        // --- in-lane online softmax for query row iw+ln ---
        float cm = fmaxf(S[0][0], S[1][0]);
#pragma unroll
        for (int r = 1; r < 8; ++r)
            cm = fmaxf(cm, fmaxf(S[0][r], S[1][r]));
        cm = fmaxf(cm, __shfl_xor(cm, 16, 32));     // merge the two lane-halves
        float mnew  = fmaxf(mrun, cm);
        float alpha = __expf(mrun - mnew);

        float rs = 0.f;
#pragma unroll
        for (int t = 0; t < 2; ++t) {
#pragma unroll
            for (int r = 0; r < 8; r += 2) {        // m = t*16 + 8h + r (consecutive pair)
                float p0 = __expf(S[t][r]     - mnew);
                float p1 = __expf(S[t][r + 1] - mnew);
                rs += p0 + p1;
                unsigned int pk = (unsigned int)f32_to_bf16(p0)
                                | ((unsigned int)f32_to_bf16(p1) << 16);
                *(unsigned int*)(Pw + ln * LROW + t * 16 + h * 8 + r) = pk;
            }
        }
        rs += __shfl_xor(rs, 16, 32);
        lrun = alpha * lrun + rs;
        mrun = mnew;

        // rescale running accumulator (alpha is lane-local scalar)
#pragma unroll
        for (int t = 0; t < 16; ++t)
#pragma unroll
            for (int r = 0; r < 8; ++r)
                Ot[t][r] *= alpha;

        // P^T B-operand (lane = column i, m = 8h+s / 16+8h+(s-8)) — once per chunk
        BF16x16 pb;
        {
            const unsigned short* prow = Pw + ln * LROW;
            pb.u[0] = *(const uint4*)(prow + h * 8);
            pb.u[1] = *(const uint4*)(prow + 16 + h * 8);
        }
        // --- O^T += V^T @ P^T over 16 channel tiles (A-load pipelined 1 ahead) ---
        BF16x16 va;
        {
            const unsigned short* vrow = bV + ln * LROW;
            va.u[0] = *(const uint4*)(vrow + h * 8);
            va.u[1] = *(const uint4*)(vrow + 16 + h * 8);
        }
#pragma unroll
        for (int t = 0; t < 16; ++t) {
            BF16x16 vn;
            if (t < 15) {
                const unsigned short* vrow = bV + ((t + 1) * 16 + ln) * LROW;
                vn.u[0] = *(const uint4*)(vrow + h * 8);
                vn.u[1] = *(const uint4*)(vrow + 16 + h * 8);
            }
            Ot[t] = __builtin_amdgcn_wmma_f32_16x16x32_bf16(
                false, va.v, false, pb.v, (short)0, Ot[t], false, false);
            va = vn;
        }
    }

    // -------- epilogue: out[b][o][i] = gamma * O^T[o][i]/l_i + x[b][o][i] ------
    // O^T is already (o,i)-major: reg r, half h -> o = t*16+r+8h; lane ln -> i.
    const float rli = 1.0f / lrun;
    const float g   = gamma[0];
#pragma unroll
    for (int t = 0; t < 16; ++t) {
#pragma unroll
        for (int r = 0; r < 8; ++r) {
            size_t idx = ((size_t)(b * OUT_CH) + t * 16 + r + 8 * h) * HN + iw + ln;
            out[idx] = g * (Ot[t][r] * rli) + x[idx];
        }
    }
}

// ---------------------------------------------------------------- launcher
extern "C" void kernel_launch(void* const* d_in, const int* in_sizes, int n_in,
                              void* d_out, int out_size, void* d_ws, size_t ws_size,
                              hipStream_t stream) {
    const float* x     = (const float*)d_in[0];
    const float* wq    = (const float*)d_in[1];
    const float* bq    = (const float*)d_in[2];
    const float* wk    = (const float*)d_in[3];
    const float* bk    = (const float*)d_in[4];
    const float* wv    = (const float*)d_in[5];
    const float* bv    = (const float*)d_in[6];
    const float* gamma = (const float*)d_in[7];
    float* out = (float*)d_out;

    // workspace: Q (1MB) | K (1MB) | Vt (8MB), all bf16
    unsigned short* Qg = (unsigned short*)d_ws;
    unsigned short* Kg = Qg + (size_t)NB * HN * DKQ;
    unsigned short* Vt = Kg + (size_t)NB * HN * DKQ;

    proj_qk_kernel<<<NB * HN / 256, 256, 0, stream>>>(x, wq, bq, wk, bk, Qg, Kg);
    proj_v_kernel<<<NB * HN / 32, 256, 0, stream>>>(x, wv, bv, Vt);
    flash_attn_kernel<<<NB * (HN / 64), 128, 0, stream>>>(Qg, Kg, Vt, x, gamma, out);
}
